// ObstacleGRU_6193342841220
// MI455X (gfx1250) — matrix operand order repaired and checked by reading, not compiled, not run on hardware
//
#include <hip/hip_runtime.h>
#include <hip/hip_fp16.h>
#include <math.h>

typedef __attribute__((ext_vector_type(16))) _Float16 v16h;
typedef __attribute__((ext_vector_type(8)))  _Float16 v8h;
typedef __attribute__((ext_vector_type(8)))  float    v8f;

#define HID   256
#define NC    5
#define HOR   30
#define FD    7
#define ROWS  64          // rows (agents) per workgroup
#define KTI   8           // HID/32 k-tiles
#define SX    264         // f16 act row stride (528B, 16B aligned)
#define STF   260         // f32 scratch row stride

// s_const layout (floats)
#define OFF_WIN1  0       // 7*256
#define OFF_BIN1  1792
#define OFF_BIN2  2048
#define OFF_LN1G  2304
#define OFF_LN1B  2560
#define OFF_BIH   2816    // 768
#define OFF_BHH   3584    // 768
#define OFF_LN2G  4352
#define OFF_LN2B  4608
#define OFF_BM1   4864    // 128
#define OFF_WM2   4992    // 128*5
#define OFF_BM2   5632    // 5
#define OFF_DELTA 5637    // 5*2
#define NCONST    5647

__device__ __forceinline__ v8f wmma_f16(v16h a, v16h b, v8f c) {
  return __builtin_amdgcn_wmma_f32_16x16x32_f16(false, a, false, b, (short)0, c, false, false);
}

// A fragment: 16x32 f16 tile, rows = rowtile*16.., from LDS row-major buffer (stride SX)
__device__ __forceinline__ v16h load_a(const _Float16* s, int rt, int kt, int lane) {
  int row = rt * 16 + (lane & 15);
  int hi  = lane >> 4;
  const _Float16* p = s + row * SX + kt * 32 + hi * 8;
  union { v16h v; v8h h[2]; } u;
  u.h[0] = *(const v8h*)p;         // K 0..7  (lane<16)  / K 8..15  (lane>=16)
  u.h[1] = *(const v8h*)(p + 16);  // K 16..23          / K 24..31
  return u.v;
}

// B fragment from pre-packed weights: block (nt,kt) holds 512 f16 in lane-major order
__device__ __forceinline__ v16h load_b(const _Float16* w, int ktiles, int nt, int kt, int lane) {
  return *(const v16h*)(w + (((nt * ktiles + kt) << 9) + (lane << 4)));
}

__device__ __forceinline__ float sigmf(float x) { return 1.0f / (1.0f + __expf(-x)); }

// Pack f32 weight W[K][N] (row-major) into WMMA-B fragment blocks (f16).
__global__ void pack_weights(const float* __restrict__ src, _Float16* __restrict__ dst,
                             int K, int N) {
  int i = blockIdx.x * blockDim.x + threadIdx.x;
  if (i >= K * N) return;
  int row = i / N, col = i % N;
  int nt = col >> 4, lo = col & 15;
  int kt = row >> 5, rs = row & 31;
  int hi = (rs >> 3) & 1,  j  = (rs & 7) + ((rs >> 4) << 3);
  int lane = lo + (hi << 4);
  int ktiles = K >> 5;
  dst[(((nt * ktiles + kt) << 9) + (lane << 4)) + j] = (_Float16)src[(size_t)row * N + col];
}

// LayerNorm over 256 cols, 4 threads/row, writes f16 (optionally ReLU'd)
__device__ __forceinline__ void layernorm_256(const float* sin, const float* g, const float* b,
                                              _Float16* out16, bool do_relu,
                                              float* s_red, int tid) {
  int q = tid & 3, row = tid >> 2;
  const float* rp = sin + row * STF + q * 64;
  float s = 0.f, s2 = 0.f;
  for (int i = 0; i < 64; ++i) { float v = rp[i]; s += v; s2 += v * v; }
  s_red[row * 8 + q]     = s;
  s_red[row * 8 + 4 + q] = s2;
  __syncthreads();
  float sum = s_red[row*8] + s_red[row*8+1] + s_red[row*8+2] + s_red[row*8+3];
  float sq  = s_red[row*8+4] + s_red[row*8+5] + s_red[row*8+6] + s_red[row*8+7];
  float mean = sum * (1.f / 256.f);
  float var  = sq * (1.f / 256.f) - mean * mean;
  float rstd = rsqrtf(var + 1e-5f);
  _Float16* op = out16 + row * SX + q * 64;
  const float* gp = g + q * 64; const float* bp = b + q * 64;
  for (int i = 0; i < 64; ++i) {
    float y = (rp[i] - mean) * rstd * gp[i] + bp[i];
    if (do_relu) y = fmaxf(y, 0.f);
    op[i] = (_Float16)y;
  }
  __syncthreads();
}

// input_proj: relu(LN(relu(feat@w_in1+b1)@w_in2+b2)) -> out16 (f16 A-layout buffer)
__device__ __forceinline__ void input_proj(const float* s_feat, const float* s_const,
                                           const _Float16* wp_in2,
                                           _Float16* s_x16, float* s_tmp32, float* s_red,
                                           _Float16* out16, int tid, int lane, int wave) {
  {  // stage A: [64,7]x[7,256] on VALU, one column per thread
    const float* w1 = s_const + OFF_WIN1;
    int col = tid;
    float wv[FD];
#pragma unroll
    for (int f = 0; f < FD; ++f) wv[f] = w1[f * 256 + col];
    float bias = s_const[OFF_BIN1 + col];
    for (int r = 0; r < ROWS; ++r) {
      float a = bias;
#pragma unroll
      for (int f = 0; f < FD; ++f) a += s_feat[r * 8 + f] * wv[f];
      s_x16[r * SX + col] = (_Float16)fmaxf(a, 0.f);
    }
  }
  __syncthreads();
  // stage B: [64,256]x[256,256] WMMA; one B-fragment feeds all 4 row-tiles
  for (int nt = wave; nt < 16; nt += 8) {
    v8f acc[4];
#pragma unroll
    for (int rt = 0; rt < 4; ++rt) acc[rt] = (v8f){0.f,0.f,0.f,0.f,0.f,0.f,0.f,0.f};
    for (int kt = 0; kt < KTI; ++kt) {
      v16h b = load_b(wp_in2, KTI, nt, kt, lane);
#pragma unroll
      for (int rt = 0; rt < 4; ++rt)
        acc[rt] = wmma_f16(load_a(s_x16, rt, kt, lane), b, acc[rt]);
    }
    int col = nt * 16 + (lane & 15);
    float bias = s_const[OFF_BIN2 + col];
#pragma unroll
    for (int rt = 0; rt < 4; ++rt) {
      int rb = rt * 16 + ((lane >> 4) << 3);
#pragma unroll
      for (int e = 0; e < 8; ++e) s_tmp32[(rb + e) * STF + col] = acc[rt][e] + bias;
    }
  }
  __syncthreads();
  layernorm_256(s_tmp32, s_const + OFF_LN1G, s_const + OFF_LN1B, out16, true, s_red, tid);
}

__global__ __launch_bounds__(256, 1) void obstacle_gru_kernel(
    const float* __restrict__ obs, const int* __restrict__ cvmask,
    const float* __restrict__ w_in1, const float* __restrict__ b_in1,
    const float* __restrict__ b_in2, const float* __restrict__ ln1g, const float* __restrict__ ln1b,
    const float* __restrict__ b_ih, const float* __restrict__ b_hh,
    const float* __restrict__ ln2g, const float* __restrict__ ln2b,
    const float* __restrict__ b_m1, const float* __restrict__ w_m2, const float* __restrict__ b_m2,
    const float* __restrict__ delta,
    const _Float16* __restrict__ wp_in2, const _Float16* __restrict__ wp_ih,
    const _Float16* __restrict__ wp_hh,  const _Float16* __restrict__ wp_m1,
    float* __restrict__ out) {
  __shared__ __attribute__((aligned(16))) _Float16 s_x16[ROWS * SX];
  __shared__ __attribute__((aligned(16))) _Float16 s_h16[ROWS * SX];
  __shared__ float s_tmp32[ROWS * STF];   // pre-LN scratch, reused by move-head
  __shared__ float s_feat[ROWS * 8];      // [..7] feat, [7] = cv mask
  __shared__ float s_red[ROWS * 8];
  __shared__ float s_const[NCONST];

  const int tid  = threadIdx.x;
  const int lane = tid & 31;
  const int wave = tid >> 5;
  const int g0   = blockIdx.x * ROWS;     // first global row

  // stage constants into LDS
  for (int i = tid; i < FD * 256; i += 256) s_const[OFF_WIN1 + i] = w_in1[i];
  for (int i = tid; i < 256; i += 256) {
    s_const[OFF_BIN1 + i] = b_in1[i]; s_const[OFF_BIN2 + i] = b_in2[i];
    s_const[OFF_LN1G + i] = ln1g[i];  s_const[OFF_LN1B + i] = ln1b[i];
    s_const[OFF_LN2G + i] = ln2g[i];  s_const[OFF_LN2B + i] = ln2b[i];
  }
  for (int i = tid; i < 768; i += 256) { s_const[OFF_BIH + i] = b_ih[i]; s_const[OFF_BHH + i] = b_hh[i]; }
  for (int i = tid; i < 128; i += 256) s_const[OFF_BM1 + i] = b_m1[i];
  for (int i = tid; i < 640; i += 256) s_const[OFF_WM2 + i] = w_m2[i];
  for (int i = tid; i < NC;  i += 256) s_const[OFF_BM2 + i] = b_m2[i];
  for (int i = tid; i < NC*2;i += 256) s_const[OFF_DELTA + i] = delta[i];
  for (int i = tid; i < ROWS * FD; i += 256) {
    int r = i / FD, f = i % FD;
    s_feat[r * 8 + f] = obs[(size_t)(g0 + r) * FD + f];
  }
  for (int i = tid; i < ROWS; i += 256) s_feat[i * 8 + 7] = (cvmask[g0 + i] != 0) ? 1.f : 0.f;
  __syncthreads();

  // h0 = input_proj(feat0)
  input_proj(s_feat, s_const, wp_in2, s_x16, s_tmp32, s_red, s_h16, tid, lane, wave);

  const v8f vzero = {0.f,0.f,0.f,0.f,0.f,0.f,0.f,0.f};

  for (int t = 0; t < HOR; ++t) {
    // x = input_proj(feat)
    input_proj(s_feat, s_const, wp_in2, s_x16, s_tmp32, s_red, s_x16, tid, lane, wave);

    // GRU gates: gi = x@w_ih, gh = h@w_hh (768 cols), fused epilogue -> s_tmp32 (pre-LN h).
    // One bundle per hid-col tile j; 6 weight fragments feed all 4 row-tiles.
    for (int j = wave; j < 16; j += 8) {
      v8f ar[4], az[4], ain[4], ahn[4];
#pragma unroll
      for (int rt = 0; rt < 4; ++rt) { ar[rt] = vzero; az[rt] = vzero; ain[rt] = vzero; ahn[rt] = vzero; }
      for (int kt = 0; kt < KTI; ++kt) {
        v16h bir = load_b(wp_ih, KTI, j,      kt, lane);
        v16h biz = load_b(wp_ih, KTI, j + 16, kt, lane);
        v16h bin_= load_b(wp_ih, KTI, j + 32, kt, lane);
        v16h bhr = load_b(wp_hh, KTI, j,      kt, lane);
        v16h bhz = load_b(wp_hh, KTI, j + 16, kt, lane);
        v16h bhn = load_b(wp_hh, KTI, j + 32, kt, lane);
#pragma unroll
        for (int rt = 0; rt < 4; ++rt) {
          v16h ax = load_a(s_x16, rt, kt, lane);
          v16h ah = load_a(s_h16, rt, kt, lane);
          ar[rt]  = wmma_f16(ax, bir, ar[rt]);
          ar[rt]  = wmma_f16(ah, bhr, ar[rt]);
          az[rt]  = wmma_f16(ax, biz, az[rt]);
          az[rt]  = wmma_f16(ah, bhz, az[rt]);
          ain[rt] = wmma_f16(ax, bin_, ain[rt]);
          ahn[rt] = wmma_f16(ah, bhn, ahn[rt]);
        }
      }
      int col = j * 16 + (lane & 15);
      float br  = s_const[OFF_BIH + col]           + s_const[OFF_BHH + col];
      float bz  = s_const[OFF_BIH + HID + col]     + s_const[OFF_BHH + HID + col];
      float bni = s_const[OFF_BIH + 2 * HID + col];
      float bnh = s_const[OFF_BHH + 2 * HID + col];
#pragma unroll
      for (int rt = 0; rt < 4; ++rt) {
        int rb = rt * 16 + ((lane >> 4) << 3);
#pragma unroll
        for (int e = 0; e < 8; ++e) {
          int row = rb + e;
          float r  = sigmf(ar[rt][e] + br);
          float zg = sigmf(az[rt][e] + bz);
          float n  = tanhf(ain[rt][e] + bni + r * (ahn[rt][e] + bnh));
          float hp = (float)s_h16[row * SX + col];
          s_tmp32[row * STF + col] = (1.f - zg) * n + zg * hp;
        }
      }
    }
    __syncthreads();
    // h = LN2(h_cand)  (f16 into s_h16)
    layernorm_256(s_tmp32, s_const + OFF_LN2G, s_const + OFF_LN2B, s_h16, false, s_red, tid);

    // move head part 1: relu(h@w_m1 + b_m1) -> s_tmp32[:, :128]; one nt per wave
    for (int nt = wave; nt < 8; nt += 8) {
      v8f acc[4];
#pragma unroll
      for (int rt = 0; rt < 4; ++rt) acc[rt] = vzero;
      for (int kt = 0; kt < KTI; ++kt) {
        v16h b = load_b(wp_m1, KTI, nt, kt, lane);
#pragma unroll
        for (int rt = 0; rt < 4; ++rt)
          acc[rt] = wmma_f16(load_a(s_h16, rt, kt, lane), b, acc[rt]);
      }
      int col = nt * 16 + (lane & 15);
      float bias = s_const[OFF_BM1 + col];
#pragma unroll
      for (int rt = 0; rt < 4; ++rt) {
        int rb = rt * 16 + ((lane >> 4) << 3);
#pragma unroll
        for (int e = 0; e < 8; ++e)
          s_tmp32[(rb + e) * STF + col] = fmaxf(acc[rt][e] + bias, 0.f);
      }
    }
    __syncthreads();

    // logits, masked store, softmax -> expected delta -> feat roll (one thread per row)
    if (tid < ROWS) {
      int row = tid;
      const float* wm2 = s_const + OFF_WM2;
      float lg[NC];
#pragma unroll
      for (int c = 0; c < NC; ++c) lg[c] = s_const[OFF_BM2 + c];
      for (int k = 0; k < 128; ++k) {
        float hv = s_tmp32[row * STF + k];
#pragma unroll
        for (int c = 0; c < NC; ++c) lg[c] += hv * wm2[k * NC + c];
      }
      float mask = s_feat[row * 8 + 7];
      size_t base = ((size_t)(g0 + row) * HOR + t) * NC;
#pragma unroll
      for (int c = 0; c < NC; ++c) out[base + c] = lg[c] * mask;
      float mx = lg[0];
#pragma unroll
      for (int c = 1; c < NC; ++c) mx = fmaxf(mx, lg[c]);
      float es[NC], se = 0.f;
#pragma unroll
      for (int c = 0; c < NC; ++c) { es[c] = __expf(lg[c] - mx); se += es[c]; }
      float inv = 1.f / se, dx = 0.f, dy = 0.f;
#pragma unroll
      for (int c = 0; c < NC; ++c) {
        float pc = es[c] * inv;
        dx += pc * s_const[OFF_DELTA + c * 2];
        dy += pc * s_const[OFF_DELTA + c * 2 + 1];
      }
      s_feat[row * 8 + 0] = fminf(fmaxf(s_feat[row * 8 + 0] + dx * mask, 0.f), 1.f);
      s_feat[row * 8 + 1] = fminf(fmaxf(s_feat[row * 8 + 1] + dy * mask, 0.f), 1.f);
    }
    __syncthreads();
  }
}

extern "C" void kernel_launch(void* const* d_in, const int* in_sizes, int n_in,
                              void* d_out, int out_size, void* d_ws, size_t ws_size,
                              hipStream_t stream) {
  const float* obs   = (const float*)d_in[0];
  const int*   cvm   = (const int*)d_in[1];
  const float* delta = (const float*)d_in[2];
  const float* w_in1 = (const float*)d_in[3];
  const float* b_in1 = (const float*)d_in[4];
  const float* w_in2 = (const float*)d_in[5];
  const float* b_in2 = (const float*)d_in[6];
  const float* ln1g  = (const float*)d_in[7];
  const float* ln1b  = (const float*)d_in[8];
  const float* w_ih  = (const float*)d_in[9];
  const float* b_ih  = (const float*)d_in[10];
  const float* w_hh  = (const float*)d_in[11];
  const float* b_hh  = (const float*)d_in[12];
  const float* ln2g  = (const float*)d_in[13];
  const float* ln2b  = (const float*)d_in[14];
  const float* w_m1  = (const float*)d_in[15];
  const float* b_m1  = (const float*)d_in[16];
  const float* w_m2  = (const float*)d_in[17];
  const float* b_m2  = (const float*)d_in[18];

  _Float16* ws     = (_Float16*)d_ws;
  _Float16* wp_in2 = ws;                               // 256*256
  _Float16* wp_ih  = ws + 65536;                       // 256*768
  _Float16* wp_hh  = ws + 65536 + 196608;              // 256*768
  _Float16* wp_m1  = ws + 65536 + 196608 + 196608;     // 256*128

  pack_weights<<<(65536  + 255) / 256, 256, 0, stream>>>(w_in2, wp_in2, 256, 256);
  pack_weights<<<(196608 + 255) / 256, 256, 0, stream>>>(w_ih,  wp_ih,  256, 768);
  pack_weights<<<(196608 + 255) / 256, 256, 0, stream>>>(w_hh,  wp_hh,  256, 768);
  pack_weights<<<(32768  + 255) / 256, 256, 0, stream>>>(w_m1,  wp_m1,  256, 128);

  obstacle_gru_kernel<<<32768 / ROWS, 256, 0, stream>>>(
      obs, cvm, w_in1, b_in1, b_in2, ln1g, ln1b, b_ih, b_hh, ln2g, ln2b,
      b_m1, w_m2, b_m2, delta, wp_in2, wp_ih, wp_hh, wp_m1, (float*)d_out);
}
